// Model_75015898792668
// MI455X (gfx1250) — compile-verified, matrix operand order checked by reading, not run on hardware
//
#include <hip/hip_runtime.h>

#define NN 100000
#define DD 128
#define HH 128
#define RR 4
#define EE 400000
#define LL 200000
#define NF (NN * 128)

typedef __attribute__((ext_vector_type(16))) __bf16 v16bf;
typedef __attribute__((ext_vector_type(8)))  float  v8f;

__device__ __forceinline__ unsigned short f2bf(float f) {
    unsigned u = __float_as_uint(f);
    u += 0x7FFFu + ((u >> 16) & 1u);          // round-to-nearest-even
    return (unsigned short)(u >> 16);
}
__device__ __forceinline__ float bf2f(unsigned short h) {
    return __uint_as_float(((unsigned)h) << 16);
}

union FragU { uint4 q[2]; v16bf v; };

__device__ __forceinline__ v16bf load_frag(const unsigned short* p, int o0, int o1) {
    FragU u;
    u.q[0] = *reinterpret_cast<const uint4*>(p + o0);   // elements 0..7
    u.q[1] = *reinterpret_cast<const uint4*>(p + o1);   // elements 8..15
    return u.v;
}

// ---------------- elementwise / prep kernels ----------------

__global__ void k_fill_f32(float* p, float v, int n) {
    int i = blockIdx.x * blockDim.x + threadIdx.x;
    if (i < n) p[i] = v;
}
__global__ void k_fill_from(float* p, const float* s, int n) {
    int i = blockIdx.x * blockDim.x + threadIdx.x;
    if (i < n) p[i] = s[0];
}
__global__ void k_f32_to_bf16(const float* in, unsigned short* out, int n) {
    int i = blockIdx.x * blockDim.x + threadIdx.x;
    if (i < n) out[i] = f2bf(in[i]);
}
// W: [R][128][128] row-major -> Wt: [R][h][k] bf16 (column-major per matrix)
__global__ void k_transpose_wl(const float* W, unsigned short* Wt) {
    int i = blockIdx.x * blockDim.x + threadIdx.x;
    if (i >= RR * 16384) return;
    int r = i >> 14, rem = i & 16383, k = rem >> 7, h = rem & 127;
    Wt[(r << 14) + (h << 7) + k] = f2bf(W[i]);
}
// Wt[h][k] = bf16( sum_r W[r][k][h] )
__global__ void k_sumtrans_wr(const float* W, unsigned short* Wt) {
    int i = blockIdx.x * blockDim.x + threadIdx.x;
    if (i >= 16384) return;
    int k = i >> 7, h = i & 127;
    float s = 0.0f;
    #pragma unroll
    for (int r = 0; r < RR; r++) s += W[r * 16384 + i];
    Wt[(h << 7) + k] = f2bf(s);
}
__global__ void k_sum_bias(const float* b, float* bs) {
    int h = threadIdx.x;
    if (h < 128) bs[h] = b[h] + b[128 + h] + b[256 + h] + b[384 + h];
}
// dW1: [256][128] -> dW1t: [h][k] (128 x 256) bf16
__global__ void k_trans_dw1(const float* W, unsigned short* Wt) {
    int i = blockIdx.x * blockDim.x + threadIdx.x;
    if (i >= 32768) return;
    int k = i >> 7, h = i & 127;
    Wt[h * 256 + k] = f2bf(W[i]);
}

// ---------------- aggregation (mean over neighbors) ----------------

__global__ void k_scatter(const unsigned short* feat, const int* src, const int* dst,
                          float* aggf, float* cnt, int nE) {
    int t = blockIdx.x * blockDim.x + threadIdx.x;   // nE*32 threads
    if (t >= nE * 32) return;
    int e = t >> 5, q = t & 31;
    int s = src[e], d = dst[e];
    const unsigned short* fp = feat + (size_t)s * 128 + q * 4;
    float* ap = aggf + (size_t)d * 128 + q * 4;
    #pragma unroll
    for (int j = 0; j < 4; j++)
        __hip_atomic_fetch_add(ap + j, bf2f(fp[j]), __ATOMIC_RELAXED, __HIP_MEMORY_SCOPE_AGENT);
    if (q == 0)
        __hip_atomic_fetch_add(cnt + d, 1.0f, __ATOMIC_RELAXED, __HIP_MEMORY_SCOPE_AGENT);
}
__global__ void k_finalize(const float* aggf, const float* cnt, unsigned short* out, int n) {
    int i = blockIdx.x * blockDim.x + threadIdx.x;
    if (i >= n) return;
    float c = cnt[i >> 7];
    out[i] = f2bf(aggf[i] * (1.0f / fmaxf(c, 1.0f)));
}

// ---------------- fused SAGE layer (WMMA) ----------------
// out[n,h] = act( sum_r agg_r[n,:]@Wl_r + self[n,:]@Wrsum + bsum ), written as bf16.
// Block = one 16-row M-tile, 8 waves cover the 128 output columns.
// All five 16x128 A-tiles are staged in LDS once; C lives in registers for all
// 20 chained WMMAs (no f32 accumulator round-trip through memory).

__global__ void __launch_bounds__(256) k_layer(const unsigned short* __restrict__ self,
                                               const unsigned short* __restrict__ aggAll,
                                               const unsigned short* __restrict__ Wlt,
                                               const unsigned short* __restrict__ Wrt,
                                               const float* __restrict__ bsum,
                                               unsigned short* __restrict__ out,
                                               int relu) {
    __shared__ unsigned short tA[5][16 * 128];          // 5 x 4 KB
    int tid = threadIdx.x;
    int mbase = blockIdx.x * 16;

    const unsigned short* srcs[5] = {
        aggAll + 0 * (size_t)NF + (size_t)mbase * 128,
        aggAll + 1 * (size_t)NF + (size_t)mbase * 128,
        aggAll + 2 * (size_t)NF + (size_t)mbase * 128,
        aggAll + 3 * (size_t)NF + (size_t)mbase * 128,
        self   + (size_t)mbase * 128,
    };
    __builtin_prefetch(self + (size_t)(mbase + 16) * 128, 0, 1);  // next M-tile
    #pragma unroll
    for (int s = 0; s < 5; s++)                          // contiguous 4 KB per tile
        ((uint4*)tA[s])[tid] = ((const uint4*)srcs[s])[tid];
    __syncthreads();

    int lane = tid & 31;
    int wave = tid >> 5;          // n-tile 0..7
    int l16  = lane & 15;
    int half = lane >> 4;
    int col  = wave * 16 + l16;

    v8f c;
    float bias = bsum[col];
    #pragma unroll
    for (int v = 0; v < 8; v++) c[v] = bias;

    #pragma unroll
    for (int s = 0; s < 5; s++) {
        const unsigned short* arow = tA[s] + l16 * 128;
        const unsigned short* brow =
            ((s < 4) ? (Wlt + (size_t)s * 16384) : Wrt) + (size_t)col * 128;
        #pragma unroll
        for (int kk = 0; kk < 4; kk++) {
            int o = kk * 32 + half * 8;
            v16bf a = load_frag(arow, o, o + 16);
            v16bf b = load_frag(brow, o, o + 16);
            c = __builtin_amdgcn_wmma_f32_16x16x32_bf16(false, a, false, b,
                                                        (short)0, c, false, false);
        }
    }

    unsigned short* obase = out + (size_t)(mbase + half * 8) * 128 + col;
    #pragma unroll
    for (int v = 0; v < 8; v++) {
        float f = c[v];
        if (relu) f = fmaxf(f, 0.0f);
        obase[(size_t)v * 128] = f2bf(f);
    }
}

// ---------------- fused edge decoder ----------------
// hidden[16 edges x 128] = relu(concat(z[row],z[col]) @ dW1 + db1); out += hidden @ dW2.
// The 16x256 concat tile is gathered into LDS once per block (8x dedup of the
// random z-row gathers); each wave does an 8-step K=256 WMMA chain, then a
// lane-shuffle reduction * dW2[col] and one f32 atomic per row per wave.

__global__ void __launch_bounds__(256) k_decoder(const unsigned short* __restrict__ zbf,
                                                 const int* __restrict__ eli,
                                                 const unsigned short* __restrict__ dW1t,
                                                 const float* __restrict__ db1,
                                                 const float* __restrict__ dW2,
                                                 float* __restrict__ out) {
    __shared__ unsigned short tZ[16 * 256];             // 8 KB concat tile
    int tid = threadIdx.x;
    int ebase = blockIdx.x * 16;

    #pragma unroll
    for (int cI = tid; cI < 512; cI += 256) {           // 512 x 16B chunks
        int row  = cI >> 5;
        int part = (cI >> 4) & 1;
        int q    = cI & 15;
        int idx  = part ? eli[LL + ebase + row] : eli[ebase + row];
        ((uint4*)tZ)[cI] = *(const uint4*)(zbf + (size_t)idx * 128 + q * 8);
    }
    __syncthreads();

    int lane = tid & 31;
    int wave = tid >> 5;
    int l16  = lane & 15;
    int half = lane >> 4;
    int col  = wave * 16 + l16;

    const unsigned short* arow = tZ + l16 * 256;
    const unsigned short* brow = dW1t + (size_t)col * 256;
    float bias = db1[col];
    v8f c;
    #pragma unroll
    for (int v = 0; v < 8; v++) c[v] = bias;

    #pragma unroll
    for (int kk = 0; kk < 8; kk++) {
        int o = kk * 32 + half * 8;
        v16bf a = load_frag(arow, o, o + 16);
        v16bf b = load_frag(brow, o, o + 16);
        c = __builtin_amdgcn_wmma_f32_16x16x32_bf16(false, a, false, b,
                                                    (short)0, c, false, false);
    }

    float scale = dW2[col];
    #pragma unroll
    for (int v = 0; v < 8; v++) {
        float p = fmaxf(c[v], 0.0f) * scale;
        p += __shfl_xor(p, 1, 32);
        p += __shfl_xor(p, 2, 32);
        p += __shfl_xor(p, 4, 32);
        p += __shfl_xor(p, 8, 32);      // sum across the 16 lanes of this half
        if (l16 == 0)
            __hip_atomic_fetch_add(out + (ebase + half * 8 + v), p,
                                   __ATOMIC_RELAXED, __HIP_MEMORY_SCOPE_AGENT);
    }
}

// ---------------- host orchestration ----------------

extern "C" void kernel_launch(void* const* d_in, const int* in_sizes, int n_in,
                              void* d_out, int out_size, void* d_ws, size_t ws_size,
                              hipStream_t stream) {
    const float* x   = (const float*)d_in[0];
    const int*   ei  = (const int*)d_in[1];
    const int*   eli = (const int*)d_in[2];
    const float* Wl1 = (const float*)d_in[3];
    const float* Wr1 = (const float*)d_in[4];
    const float* b1  = (const float*)d_in[5];
    const float* Wl2 = (const float*)d_in[6];
    const float* Wr2 = (const float*)d_in[7];
    const float* b2  = (const float*)d_in[8];
    const float* dW1 = (const float*)d_in[9];
    const float* db1 = (const float*)d_in[10];
    const float* dW2 = (const float*)d_in[11];
    const float* db2 = (const float*)d_in[12];
    float* out = (float*)d_out;

    char* ws = (char*)d_ws;
    size_t off = 0;
    auto alloc = [&](size_t bytes) -> char* {
        char* p = ws + off;
        off += (bytes + 255) & ~(size_t)255;
        return p;
    };
    unsigned short* xbf    = (unsigned short*)alloc((size_t)NF * 2);
    unsigned short* hbf    = (unsigned short*)alloc((size_t)NF * 2);
    unsigned short* zbf    = (unsigned short*)alloc((size_t)NF * 2);
    unsigned short* aggAll = (unsigned short*)alloc((size_t)RR * NF * 2);
    float* aggf = (float*)alloc((size_t)NF * 4);
    float* cnt  = (float*)alloc((size_t)NN * 4);
    unsigned short* Wl1t = (unsigned short*)alloc(RR * 16384 * 2);
    unsigned short* Wr1t = (unsigned short*)alloc(16384 * 2);
    unsigned short* Wl2t = (unsigned short*)alloc(RR * 16384 * 2);
    unsigned short* Wr2t = (unsigned short*)alloc(16384 * 2);
    unsigned short* dW1t = (unsigned short*)alloc(32768 * 2);
    float* bs1 = (float*)alloc(128 * 4);
    float* bs2 = (float*)alloc(128 * 4);

    auto nb = [](long long n) { return (unsigned)((n + 255) / 256); };

    // ---- weight / input prep ----
    k_f32_to_bf16<<<nb(NF), 256, 0, stream>>>(x, xbf, NF);
    k_transpose_wl<<<nb(RR * 16384), 256, 0, stream>>>(Wl1, Wl1t);
    k_transpose_wl<<<nb(RR * 16384), 256, 0, stream>>>(Wl2, Wl2t);
    k_sumtrans_wr<<<nb(16384), 256, 0, stream>>>(Wr1, Wr1t);
    k_sumtrans_wr<<<nb(16384), 256, 0, stream>>>(Wr2, Wr2t);
    k_sum_bias<<<1, 128, 0, stream>>>(b1, bs1);
    k_sum_bias<<<1, 128, 0, stream>>>(b2, bs2);
    k_trans_dw1<<<nb(32768), 256, 0, stream>>>(dW1, dW1t);

    // ---- layer 1 ----
    for (int r = 0; r < RR; r++) {
        const int* src = ei + (size_t)r * 2 * EE;
        const int* dst = src + EE;
        k_fill_f32<<<nb(NF), 256, 0, stream>>>(aggf, 0.0f, NF);
        k_fill_f32<<<nb(NN), 256, 0, stream>>>(cnt, 0.0f, NN);
        k_scatter<<<nb((long long)EE * 32), 256, 0, stream>>>(xbf, src, dst, aggf, cnt, EE);
        k_finalize<<<nb(NF), 256, 0, stream>>>(aggf, cnt, aggAll + (size_t)r * NF, NF);
    }
    k_layer<<<NN / 16, 256, 0, stream>>>(xbf, aggAll, Wl1t, Wr1t, bs1, hbf, 1);

    // ---- layer 2 ----
    for (int r = 0; r < RR; r++) {
        const int* src = ei + (size_t)r * 2 * EE;
        const int* dst = src + EE;
        k_fill_f32<<<nb(NF), 256, 0, stream>>>(aggf, 0.0f, NF);
        k_fill_f32<<<nb(NN), 256, 0, stream>>>(cnt, 0.0f, NN);
        k_scatter<<<nb((long long)EE * 32), 256, 0, stream>>>(hbf, src, dst, aggf, cnt, EE);
        k_finalize<<<nb(NF), 256, 0, stream>>>(aggf, cnt, aggAll + (size_t)r * NF, NF);
    }
    k_layer<<<NN / 16, 256, 0, stream>>>(hbf, aggAll, Wl2t, Wr2t, bs2, zbf, 0);

    // ---- decoder ----
    k_fill_from<<<nb(LL), 256, 0, stream>>>(out, db2, LL);
    k_decoder<<<LL / 16, 256, 0, stream>>>(zbf, eli, dW1t, db1, dW2, out);

    (void)in_sizes; (void)n_in; (void)out_size; (void)ws_size;
}
